// Wavefunction_48627619726081
// MI455X (gfx1250) — matrix operand order repaired and checked by reading, not compiled, not run on hardware
//
#include <hip/hip_runtime.h>
#include <math.h>

// Fully fused per-batch-element wavefunction evaluation for MI455X (gfx1250).
// One 288-thread workgroup (9 wave32) per batch element; all activations live
// in LDS (~56 KB), all 64-wide layers run as v_wmma_f32_16x16x32_bf16 with
// f32 accumulation. A one-time prep kernel pre-transposes/converts all WMMA
// weights to bf16 in d_ws so per-block staging is a pure b128 copy.

typedef __bf16 bf16;
typedef __attribute__((ext_vector_type(16))) __bf16 v16bf;
typedef __attribute__((ext_vector_type(8)))  __bf16 v8bf;
typedef __attribute__((ext_vector_type(8)))  float  v8f;

#define NT 288          // 9 waves of 32
#define NPART 12
#define NEDGE 144
#define SQWC 1.0f       // sqrt(OMEGA), OMEGA=1
#define NSLOT 15        // pre-converted 64x64 bf16 W^T matrices in d_ws

struct WFParams { const float* p[48]; const bf16* wt; float* out; };
struct PrepParams { const float* w[NSLOT]; int off[NSLOT]; bf16* dst; };

__device__ __forceinline__ float geluf(float x) {
  return 0.5f * x * (1.0f + erff(x * 0.70710678118654752f));
}

__device__ __forceinline__ v16bf cat8(v8bf a, v8bf b) {
  return __builtin_shufflevector(a, b, 0,1,2,3,4,5,6,7,8,9,10,11,12,13,14,15);
}

// One 16-row x 64-col tile of A[16,64] @ W[64,64], bf16 in LDS, f32 accum.
// A: row-major [row][64] bf16, base already offset to tile row 0.
// Wt: W transposed, [out][in] bf16 (row = output column, 64 K entries).
// ISA frag layouts: A lane ln holds row ln%16, halves = K {kc*32+hi*8..+7,
// kc*32+16+hi*8..+7}; B lane ln holds col ln%16, same K order as A.
__device__ __forceinline__ void gemm_tile(const bf16* A, const bf16* Wt,
                                          v8f acc[4], int lane) {
  const int cm = lane & 15, hi = lane >> 4;
  const bf16* arow = A + cm * 64;
  v16bf a0 = cat8(*(const v8bf*)(arow +      hi * 8), *(const v8bf*)(arow + 16 + hi * 8));
  v16bf a1 = cat8(*(const v8bf*)(arow + 32 + hi * 8), *(const v8bf*)(arow + 48 + hi * 8));
#pragma unroll
  for (int t = 0; t < 4; ++t) {
    const bf16* wrow = Wt + (t * 16 + cm) * 64;
    v16bf b0 = cat8(*(const v8bf*)(wrow +      hi * 8), *(const v8bf*)(wrow + 16 + hi * 8));
    v16bf b1 = cat8(*(const v8bf*)(wrow + 32 + hi * 8), *(const v8bf*)(wrow + 48 + hi * 8));
    acc[t] = __builtin_amdgcn_wmma_f32_16x16x32_bf16(false, a0, false, b0, (short)0, acc[t], false, false);
    acc[t] = __builtin_amdgcn_wmma_f32_16x16x32_bf16(false, a1, false, b1, (short)0, acc[t], false, false);
  }
}

// Copy a pre-converted bf16 W^T (8 KB) from d_ws into LDS + f32 bias.
__device__ __forceinline__ void stage_ws(const bf16* src, const float* b,
                                         bf16* sW, float* sBias, int tid) {
  const uint4* s = (const uint4*)src;
  uint4* d = (uint4*)sW;
  for (int idx = tid; idx < 512; idx += NT) d[idx] = s[idx];
  for (int c = tid; c < 64; c += NT) sBias[c] = b[c];
}

// One-time weight transpose + f32->bf16 conversion into d_ws.
__global__ __launch_bounds__(256) void prep_kernel(PrepParams PP) {
  const int slot = blockIdx.x;
  const float* W = PP.w[slot];
  const int off = PP.off[slot];
  bf16* dst = PP.dst + slot * 4096;
  for (int idx = threadIdx.x; idx < 4096; idx += 256) {
    int in = idx >> 6, out = idx & 63;
    dst[out * 64 + in] = (bf16)W[(in + off) * 64 + out];
  }
}

__global__ __launch_bounds__(NT) void wf_kernel(WFParams P) {
  __shared__ __align__(16) bf16 sA[NEDGE * 64];    // activation ping
  __shared__ __align__(16) bf16 sB[NEDGE * 64];    // activation pong
  __shared__ __align__(16) bf16 sW[64 * 64];       // W^T bf16
  __shared__ __align__(16) bf16 sHVb[16 * 64];     // h_v bf16 (rows 12..15 = 0)
  __shared__ float sHV[NPART * 64];                // h_v f32 (residual)
  __shared__ float sP[NPART * 64];                 // h_v @ W_top
  __shared__ float sAcc[NPART * 64];               // e2v scatter accumulator
  __shared__ float sBias[64];
  __shared__ float sXS[NPART * 2], sX[NPART * 2], sDX[NPART * 2];
  __shared__ float sPhiHO[NPART * 6];
  __shared__ float sJ[16];                         // 0-4 phi, 5-9 psi, 10 cusp,
                                                   // 11-12 logdets, 14-15 dx mean
  __shared__ float sH64[64];

  const int b = blockIdx.x;
  const int tid = threadIdx.x;
  const int lane = tid & 31;
  const int wave = tid >> 5;
  const float* x = P.p[0] + b * NPART * 2;
  const float* spin = P.p[47];

  for (int i = tid; i < NPART * 2; i += NT) { float v = x[i]; sX[i] = v; sXS[i] = v * SQWC; }
  for (int i = tid; i < NPART * 64; i += NT) sAcc[i] = 0.0f;
  if (tid < 16) sJ[tid] = 0.0f;
  __syncthreads();

  // ---------------- backflow: node MLP ----------------
  {
    const float* W0 = P.p[1]; const float* b0 = P.p[2];
    for (int idx = tid; idx < 16 * 64; idx += NT) {     // rows 12..15 zero pad
      int pi = idx >> 6, c = idx & 63; float val = 0.0f;
      if (pi < NPART) {
        val = geluf(sXS[pi*2]*W0[c] + sXS[pi*2+1]*W0[64+c] + spin[pi]*W0[128+c] + b0[c]);
      }
      sA[idx] = (bf16)val;
    }
    stage_ws(P.wt + 0 * 4096, P.p[4], sW, sBias, tid);
  }
  __syncthreads();
  if (wave == 0) {                                      // node layer 2 (WMMA)
    v8f acc[4] = {};
    gemm_tile(sA, sW, acc, lane);
    const int cm = lane & 15, hi = lane >> 4;
#pragma unroll
    for (int t = 0; t < 4; ++t)
#pragma unroll
      for (int v = 0; v < 8; ++v) {
        int row = v + 8 * hi, col = t * 16 + cm;
        float val = (row < NPART) ? (acc[t][v] + sBias[col]) : 0.0f;
        if (row < NPART) sHV[row * 64 + col] = val;
        sHVb[row * 64 + col] = (bf16)val;
      }
  }
  __syncthreads();

  // ---------------- backflow: edge MLP ----------------
  {
    const float* W0 = P.p[5]; const float* b0 = P.p[6];
    for (int e = tid; e < NEDGE; e += NT) {
      int i = e / NPART, j = e % NPART;
      float d0 = sXS[j*2] - sXS[i*2], d1 = sXS[j*2+1] - sXS[i*2+1];
      float r2 = d0*d0 + d1*d1, rr = sqrtf(r2 + 1e-12f);
      for (int c = 0; c < 64; ++c) {
        float v = d0*W0[c] + d1*W0[64+c] + rr*W0[128+c] + r2*W0[192+c] + b0[c];
        sA[e * 64 + c] = (bf16)geluf(v);
      }
    }
    stage_ws(P.wt + 1 * 4096, P.p[8], sW, sBias, tid);
  }
  __syncthreads();
  {                                                     // edge layer 2 -> h_e (sB)
    int r0 = wave * 16;
    v8f acc[4] = {};
    gemm_tile(sA + r0 * 64, sW, acc, lane);
    const int cm = lane & 15, hi = lane >> 4;
#pragma unroll
    for (int t = 0; t < 4; ++t)
#pragma unroll
      for (int v = 0; v < 8; ++v) {
        int row = r0 + v + 8 * hi, col = t * 16 + cm;
        sB[row * 64 + col] = (bf16)(acc[t][v] + sBias[col]);
      }
  }
  __syncthreads();

  // ---------------- message-passing layers ----------------
  for (int l = 0; l < 2; ++l) {
    const int sl = 2 + l * 5;                           // slot base for this layer
    const float* bv = P.p[10 + l * 4];
    stage_ws(P.wt + (sl + 0) * 4096, bv, sW, sBias, tid);   // W_top (h_v part)
    __syncthreads();
    if (wave == 0) {                                    // P = h_v @ W_top
      v8f acc[4] = {};
      gemm_tile(sHVb, sW, acc, lane);
      const int cm = lane & 15, hi = lane >> 4;
#pragma unroll
      for (int t = 0; t < 4; ++t)
#pragma unroll
        for (int v = 0; v < 8; ++v) {
          int row = v + 8 * hi, col = t * 16 + cm;
          if (row < NPART) sP[row * 64 + col] = acc[t][v];
        }
    }
    __syncthreads();
    stage_ws(P.wt + (sl + 1) * 4096, bv, sW, sBias, tid);   // W_bot (h_e part)
    __syncthreads();
    {                                                   // v2e layer1: gelu(he@Wb + P[i] + b)
      int r0 = wave * 16;
      v8f acc[4] = {};
      gemm_tile(sB + r0 * 64, sW, acc, lane);
      const int cm = lane & 15, hi = lane >> 4;
#pragma unroll
      for (int t = 0; t < 4; ++t)
#pragma unroll
        for (int v = 0; v < 8; ++v) {
          int row = r0 + v + 8 * hi, col = t * 16 + cm;
          int i = row / NPART;
          sA[row * 64 + col] = (bf16)geluf(acc[t][v] + sP[i * 64 + col] + sBias[col]);
        }
    }
    __syncthreads();
    stage_ws(P.wt + (sl + 2) * 4096, P.p[12 + l * 4], sW, sBias, tid);
    __syncthreads();
    {                                                   // v2e layer2 -> new h_e (sB)
      int r0 = wave * 16;
      v8f acc[4] = {};
      gemm_tile(sA + r0 * 64, sW, acc, lane);
      const int cm = lane & 15, hi = lane >> 4;
#pragma unroll
      for (int t = 0; t < 4; ++t)
#pragma unroll
        for (int v = 0; v < 8; ++v) {
          int row = r0 + v + 8 * hi, col = t * 16 + cm;
          sB[row * 64 + col] = (bf16)(acc[t][v] + sBias[col]);
        }
    }
    __syncthreads();
    stage_ws(P.wt + (sl + 3) * 4096, P.p[18 + l * 4], sW, sBias, tid);
    __syncthreads();
    {                                                   // e2v layer1 (gelu)
      int r0 = wave * 16;
      v8f acc[4] = {};
      gemm_tile(sB + r0 * 64, sW, acc, lane);
      const int cm = lane & 15, hi = lane >> 4;
#pragma unroll
      for (int t = 0; t < 4; ++t)
#pragma unroll
        for (int v = 0; v < 8; ++v) {
          int row = r0 + v + 8 * hi, col = t * 16 + cm;
          sA[row * 64 + col] = (bf16)geluf(acc[t][v] + sBias[col]);
        }
    }
    __syncthreads();
    stage_ws(P.wt + (sl + 4) * 4096, P.p[20 + l * 4], sW, sBias, tid);
    __syncthreads();
    {                                                   // e2v layer2 + mask + scatter-sum
      int r0 = wave * 16;
      v8f acc[4] = {};
      gemm_tile(sA + r0 * 64, sW, acc, lane);
      const int cm = lane & 15, hi = lane >> 4;
#pragma unroll
      for (int t = 0; t < 4; ++t)
#pragma unroll
        for (int v = 0; v < 8; ++v) {
          int row = r0 + v + 8 * hi, col = t * 16 + cm;
          int i = row / NPART, j = row % NPART;
          float val = acc[t][v] + sBias[col];
          if (i != j) atomicAdd(&sAcc[j * 64 + col], val);   // ds_add_f32
        }
    }
    __syncthreads();
    for (int idx = tid; idx < NPART * 64; idx += NT) {  // h_v residual update
      float nv = sHV[idx] + sAcc[idx] * (1.0f / 11.0f);
      sHV[idx] = nv;
      sHVb[idx] = (bf16)nv;                             // pad rows stay 0
      sAcc[idx] = 0.0f;
    }
    __syncthreads();
  }

  // ---------------- head: dx ----------------
  stage_ws(P.wt + 12 * 4096, P.p[26], sW, sBias, tid);
  __syncthreads();
  if (wave == 0) {                                      // head layer1 (tanh)
    v8f acc[4] = {};
    gemm_tile(sHVb, sW, acc, lane);
    const int cm = lane & 15, hi = lane >> 4;
#pragma unroll
    for (int t = 0; t < 4; ++t)
#pragma unroll
      for (int v = 0; v < 8; ++v) {
        int row = v + 8 * hi, col = t * 16 + cm;
        float val = (row < NPART) ? tanhf(acc[t][v] + sBias[col]) : 0.0f;
        sA[row * 64 + col] = (bf16)val;
      }
  }
  __syncthreads();
  {
    const float* W1 = P.p[27]; const float* b1 = P.p[28];
    float sp = log1pf(expf(P.p[29][0]));                // softplus(scale)
    for (int idx = tid; idx < NPART * 2; idx += NT) {
      int pi = idx >> 1, d = idx & 1;
      float s = b1[d];
      for (int k = 0; k < 64; ++k) s += (float)sA[pi * 64 + k] * W1[k * 2 + d];
      sDX[idx] = s * sp;
    }
  }
  __syncthreads();
  if (tid < 2) {
    float m = 0.0f;
    for (int pi = 0; pi < NPART; ++pi) m += sDX[pi * 2 + tid];
    sJ[14 + tid] = m / (float)NPART;
  }
  __syncthreads();

  // ---------------- Slater: HO basis + slogdet ----------------
  for (int pi = tid; pi < NPART; pi += NT) {
    float ux = SQWC * (sX[pi*2]   + sDX[pi*2]   - sJ[14]);
    float uy = SQWC * (sX[pi*2+1] + sDX[pi*2+1] - sJ[15]);
    float env = expf(-0.5f * (ux*ux + uy*uy));
    float hx1 = 2.0f*ux, hy1 = 2.0f*uy, hx2 = 4.0f*ux*ux - 2.0f, hy2 = 4.0f*uy*uy - 2.0f;
    sPhiHO[pi*6+0] = env;        sPhiHO[pi*6+1] = hx1*env;     sPhiHO[pi*6+2] = hy1*env;
    sPhiHO[pi*6+3] = hx2*env;    sPhiHO[pi*6+4] = hx1*hy1*env; sPhiHO[pi*6+5] = hy2*env;
  }
  __syncthreads();
  if (tid < 2) {                                        // LU slogdet of 6x6 (up / dn)
    const float* C = P.p[46];
    float a[6][6];
    for (int r = 0; r < 6; ++r)
      for (int c = 0; c < 6; ++c) {
        float s = 0.0f;
        for (int k = 0; k < 6; ++k) s += sPhiHO[(tid * 6 + r) * 6 + k] * C[k * 6 + c];
        a[r][c] = s;
      }
    float ld = 0.0f;
    for (int k = 0; k < 6; ++k) {
      int piv = k; float mx = fabsf(a[k][k]);
      for (int r = k + 1; r < 6; ++r) { float m = fabsf(a[r][k]); if (m > mx) { mx = m; piv = r; } }
      if (piv != k) for (int c = 0; c < 6; ++c) { float t = a[k][c]; a[k][c] = a[piv][c]; a[piv][c] = t; }
      float akk = a[k][k];
      ld += logf(fabsf(akk));
      for (int r = k + 1; r < 6; ++r) {
        float f = a[r][k] / akk;
        for (int c = k; c < 6; ++c) a[r][c] -= f * a[k][c];
      }
    }
    sJ[11 + tid] = ld;
  }

  // ---------------- Jastrow: phi branch ----------------
  {
    const float* W0 = P.p[30]; const float* b0 = P.p[31];
    for (int idx = tid; idx < 16 * 64; idx += NT) {
      int pi = idx >> 6, c = idx & 63; float val = 0.0f;
      if (pi < NPART) {
        float x0 = sXS[pi*2], x1 = sXS[pi*2+1], r2 = x0*x0 + x1*x1;
        val = geluf(x0*W0[c] + x1*W0[64+c] + r2*W0[128+c] + b0[c]);
      }
      sA[idx] = (bf16)val;
    }
    stage_ws(P.wt + 13 * 4096, P.p[33], sW, sBias, tid);
  }
  __syncthreads();
  if (wave == 0) {                                      // phi layer2 (WMMA, gelu)
    v8f acc[4] = {};
    gemm_tile(sA, sW, acc, lane);
    const int cm = lane & 15, hi = lane >> 4;
#pragma unroll
    for (int t = 0; t < 4; ++t)
#pragma unroll
      for (int v = 0; v < 8; ++v) {
        int row = v + 8 * hi, col = t * 16 + cm;
        float val = (row < NPART) ? geluf(acc[t][v] + sBias[col]) : 0.0f;
        sB[row * 64 + col] = (bf16)val;
      }
  }
  __syncthreads();
  {                                                     // phi layer3 (64->5) + mean; psi layer1
    const float* W2 = P.p[34]; const float* b2 = P.p[35];
    for (int idx = tid; idx < NPART * 5; idx += NT) {
      int pi = idx / 5, d = idx % 5;
      float s = b2[d];
      for (int k = 0; k < 64; ++k) s += (float)sB[pi * 64 + k] * W2[k * 5 + d];
      atomicAdd(&sJ[d], s * (1.0f / (float)NPART));
    }
    const float* W0 = P.p[36]; const float* b0 = P.p[37];
    for (int e = tid; e < 66; e += NT) {
      int t = e, i = 0;
      while (t >= 11 - i) { t -= 11 - i; ++i; }
      int j = i + 1 + t;
      float d0 = sX[i*2] - sX[j*2], d1 = sX[i*2+1] - sX[j*2+1];
      float r = sqrtf(d0*d0 + d1*d1 + 1e-12f);
      float f0 = log1pf(r), f1 = r / (1.0f + r), f2 = expf(-r*r);
      float f3 = expf(-0.5f*r), f4 = expf(-r), f5 = expf(-2.0f*r);
      for (int c = 0; c < 64; ++c) {
        float v = f0*W0[c] + f1*W0[64+c] + f2*W0[128+c] + f3*W0[192+c] + f4*W0[256+c] + f5*W0[320+c] + b0[c];
        sA[e * 64 + c] = (bf16)geluf(v);
      }
      atomicAdd(&sJ[10], r * f4);                        // cusp, gamma = 1/(D-1) = 1
    }
    for (int idx = 66 * 64 + tid; idx < 80 * 64; idx += NT) sA[idx] = (bf16)0.0f;
    stage_ws(P.wt + 14 * 4096, P.p[39], sW, sBias, tid);
  }
  __syncthreads();
  if (wave < 5) {                                       // psi layer2 (WMMA, gelu): 66 rows -> 5 tiles
    int r0 = wave * 16;
    v8f acc[4] = {};
    gemm_tile(sA + r0 * 64, sW, acc, lane);
    const int cm = lane & 15, hi = lane >> 4;
#pragma unroll
    for (int t = 0; t < 4; ++t)
#pragma unroll
      for (int v = 0; v < 8; ++v) {
        int row = r0 + v + 8 * hi, col = t * 16 + cm;
        float val = (row < 66) ? geluf(acc[t][v] + sBias[col]) : 0.0f;
        sB[row * 64 + col] = (bf16)val;
      }
  }
  __syncthreads();
  {                                                     // psi layer3 (64->5) + mean
    const float* W2 = P.p[40]; const float* b2 = P.p[41];
    for (int idx = tid; idx < 66 * 5; idx += NT) {
      int e = idx / 5, d = idx % 5;
      float s = b2[d];
      for (int k = 0; k < 64; ++k) s += (float)sB[e * 64 + k] * W2[k * 5 + d];
      atomicAdd(&sJ[5 + d], s * (1.0f / 66.0f));
    }
  }
  __syncthreads();
  {                                                     // rho layer1 (10->64, gelu)
    const float* W0 = P.p[42]; const float* b0 = P.p[43];
    for (int c = tid; c < 64; c += NT) {
      float s = b0[c];
      for (int k = 0; k < 10; ++k) s += sJ[k] * W0[k * 64 + c];
      sH64[c] = geluf(s);
    }
  }
  __syncthreads();
  if (tid == 0) {                                       // rho layer2 + combine
    const float* W1 = P.p[44];
    float f = P.p[45][0];
    for (int k = 0; k < 64; ++k) f += sH64[k] * W1[k];
    P.out[b] = sJ[11] + sJ[12] + f + sJ[10];
  }
}

extern "C" void kernel_launch(void* const* d_in, const int* in_sizes, int n_in,
                              void* d_out, int out_size, void* d_ws, size_t ws_size,
                              hipStream_t stream) {
  (void)ws_size; (void)out_size;
  WFParams P;
  for (int i = 0; i < 48 && i < n_in; ++i) P.p[i] = (const float*)d_in[i];
  P.wt = (const bf16*)d_ws;
  P.out = (float*)d_out;

  // Slot map: 0 nodeW1, 1 edgeW1, per-layer l: 2+5l..6+5l = {v2e Wtop, v2e Wbot,
  // v2e W1, e2v W0, e2v W1}, 12 headW0, 13 phiW1, 14 psiW1.
  PrepParams PP;
  PP.dst = (bf16*)d_ws;
  const int widx[NSLOT] = {3, 7, 9, 9, 11, 17, 19, 13, 13, 15, 21, 23, 25, 32, 38};
  const int woff[NSLOT] = {0, 0, 0, 64, 0, 0, 0, 0, 64, 0, 0, 0, 0, 0, 0};
  for (int s = 0; s < NSLOT; ++s) {
    PP.w[s] = (const float*)d_in[widx[s]];
    PP.off[s] = woff[s];
  }

  int B = in_sizes[0] / (NPART * 2);                    // 4096
  prep_kernel<<<dim3(NSLOT), dim3(256), 0, stream>>>(PP);
  wf_kernel<<<dim3(B), dim3(NT), 0, stream>>>(P);
}
